// STEncoder_HIN_76338748719595
// MI455X (gfx1250) — compile-verified, matrix operand order checked by reading, not run on hardware
//
#include <hip/hip_runtime.h>
#include <hip/hip_bf16.h>
#include <math.h>

typedef __attribute__((ext_vector_type(16))) _Float16 v16h;
typedef __attribute__((ext_vector_type(8)))  float    v8f;
typedef __attribute__((ext_vector_type(2)))  float    v2f;

#define TT   8
#define NN   768
#define RR   3
#define DD   32
#define NHEADS 8
#define HD   4
#define HID  2048

// ---------------------------------------------------------------------------
// Fragment loader (wave32, 16-bit A/B layout per CDNA5 ISA 7.12.2).
// For a row-major f32 matrix it produces, per lane, the K-striped v16h:
//   lanes 0-15 : row r0+lane,    elems 0..7 = K[0..7],  8..15 = K[16..23]
//   lanes 16-31: row r0+lane-16, elems 0..7 = K[8..15], 8..15 = K[24..31]
// This striping is identical for A fragments (lane=row M) and B fragments
// (lane=col N, reading Bt row-major), so one loader serves both.
// ---------------------------------------------------------------------------
__device__ __forceinline__ v16h load_frag(const float* __restrict__ p,
                                          int r0, int k0, int ld, int lane) {
  const float* q = p + (size_t)(r0 + (lane & 15)) * ld + k0 + ((lane >> 4) << 3);
  v16h a;
#pragma unroll
  for (int i = 0; i < 8; ++i) {
    a[i]     = (_Float16)q[i];
    a[i + 8] = (_Float16)q[16 + i];
  }
  return a;
}

// ---------------------------------------------------------------------------
// Weight pre-swizzle: W (N,K) f32 row-major -> fragment-ordered f16:
//   [tn][kk][lane][16 halves]; one contiguous 32-byte load per lane.
// ---------------------------------------------------------------------------
__global__ void k_swz(const float* __restrict__ W, _Float16* __restrict__ out,
                      int N, int K) {
  int idx = blockIdx.x * blockDim.x + threadIdx.x;
  int kt = K >> 5;
  int total = (N >> 4) * kt * 32;
  if (idx >= total) return;
  int lane = idx & 31;
  int frag = idx >> 5;               // tn*kt + kk
  int tn = frag / kt, kk = frag % kt;
  const float* q = W + (size_t)(tn * 16 + (lane & 15)) * K + kk * 32 +
                   ((lane >> 4) << 3);
  _Float16* o = out + (size_t)idx * 16;
#pragma unroll
  for (int i = 0; i < 8; ++i) {
    o[i]     = (_Float16)q[i];
    o[i + 8] = (_Float16)q[16 + i];
  }
}

// ---------------------------------------------------------------------------
// Generic GEMM computed transposed:  Ct(16xN-col tile, 16 act rows) =
// Wtile(16xK) @ At(Kx16).  Pre-swizzled W is the A operand; activations load
// with load_frag as the B operand.  Result: lane L holds out-cols
// tn*16 + 8*(L>=16) + v for act row tm*16 + (L&15)  -> each lane's 8 outputs
// are CONTIGUOUS in row-major C, and the bias run is contiguous too.
// ---------------------------------------------------------------------------
template <int ACT>
__global__ void k_gemm(const float* __restrict__ A, const _Float16* __restrict__ Ws,
                       const float* __restrict__ bias, float* __restrict__ C,
                       int M, int N, int K) {
  int lane = threadIdx.x & 31, wv = threadIdx.x >> 5;
  int tile = blockIdx.x * 4 + wv;
  int tn_n = N >> 4;
  int total = (M >> 4) * tn_n;
  if (tile >= total) return;               // wave-uniform exit (EXEC stays full)
  int tm = tile / tn_n, tn = tile % tn_n;
  int ktiles = K >> 5;
  const v16h* Wf = (const v16h*)Ws;
  v8f c = {};
  for (int kk = 0; kk < ktiles; ++kk) {
    v16h aw = Wf[((size_t)tn * ktiles + kk) * 32 + lane];
    v16h bx = load_frag(A, tm * 16, kk * 32, K, lane);
    c = __builtin_amdgcn_wmma_f32_16x16x32_f16(false, aw, false, bx, (short)0, c,
                                               false, false);
  }
  int nn = lane & 15, off8 = (lane >> 4) << 3;
  const float* bb = bias + tn * 16 + off8;
  float* cp = C + (size_t)(tm * 16 + nn) * N + tn * 16 + off8;
#pragma unroll
  for (int v = 0; v < 8; ++v) {
    float val = c[v] + bb[v];
    if (ACT) val = fmaxf(val, 0.0f);
    cp[v] = val;
  }
}

// ---------------------------------------------------------------------------
// Fused FFN, LDS-free:  D = relu(X@W1^T + b1) @ W2^T + b2
// First GEMM is computed transposed (Ht = W1chunk @ Xt): its two C tiles land
// in exactly the A-fragment striping the second GEMM needs, so bias+ReLU+cvt
// happen in registers and feed the accumulating WMMAs directly.  Hidden
// activations never exist outside VGPRs.
// ---------------------------------------------------------------------------
__global__ void k_ffn(const float* __restrict__ X, const _Float16* __restrict__ W1s,
                      const float* __restrict__ B1, const _Float16* __restrict__ W2s,
                      const float* __restrict__ B2, float* __restrict__ D, int M) {
  int lane = threadIdx.x & 31, wv = threadIdx.x >> 5;
  int rb = blockIdx.x * 4 + wv;
  if (rb * 16 >= M) return;
  int row0 = rb * 16;
  v16h xb = load_frag(X, row0, 0, 32, lane);   // B fragment of X^T (loaded once)
  v8f c0 = {}, c1 = {};
  int off8 = (lane >> 4) << 3;
  int nn = lane & 15;
  const v16h* W1f = (const v16h*)W1s;   // [tn 0..127][lane]
  const v16h* W2f = (const v16h*)W2s;   // [tn 0..1][kk 0..63][lane]
  for (int hk = 0, kt = 0; hk < HID; hk += 32, ++kt) {
    v16h aw0 = W1f[(size_t)((hk >> 4) + 0) * 32 + lane];
    v16h aw1 = W1f[(size_t)((hk >> 4) + 1) * 32 + lane];
    v8f h0 = {}, h1 = {};
    h0 = __builtin_amdgcn_wmma_f32_16x16x32_f16(false, aw0, false, xb, (short)0,
                                                h0, false, false);
    h1 = __builtin_amdgcn_wmma_f32_16x16x32_f16(false, aw1, false, xb, (short)0,
                                                h1, false, false);
    // lane L holds H[xrow=L&15, hid=hk + v + 8*(L>=16)] (tile0), +16 (tile1):
    // exactly the A-fragment striping for the second GEMM.
    const float* bp0 = B1 + hk + off8;
    const float* bp1 = B1 + hk + 16 + off8;
    v16h ha;
#pragma unroll
    for (int i = 0; i < 8; ++i) {
      ha[i]     = (_Float16)fmaxf(h0[i] + bp0[i], 0.0f);
      ha[i + 8] = (_Float16)fmaxf(h1[i] + bp1[i], 0.0f);
    }
    v16h b0 = W2f[(size_t)kt * 32 + lane];
    v16h b1 = W2f[(size_t)(64 + kt) * 32 + lane];
    c0 = __builtin_amdgcn_wmma_f32_16x16x32_f16(false, ha, false, b0, (short)0,
                                                c0, false, false);
    c1 = __builtin_amdgcn_wmma_f32_16x16x32_f16(false, ha, false, b1, (short)0,
                                                c1, false, false);
  }
#pragma unroll
  for (int v = 0; v < 8; ++v) {
    D[(size_t)(row0 + off8 + v) * 32 + nn]      = c0[v] + B2[nn];
    D[(size_t)(row0 + off8 + v) * 32 + 16 + nn] = c1[v] + B2[16 + nn];
  }
}

// ---------------------------------------------------------------------------
// Masked spatial attention, L=768, per-head head-dim 4.
// grid = (48 row-blocks, 8 heads), block = 256 (8 waves).
// Scores via v_wmma_f32_16x16x4_f32 (K=4 == head dim, exact f32), additive
// mask rebuilt from int nei (incl. the +eye "==2 stays 2" quirk), two-pass
// softmax over a 16x768 LDS strip, P@V via lane-split VALU reduction.
// qkv layout: row l, cols [0:32)=q [32:64)=k [64:96)=v, head h at h*4.
// ---------------------------------------------------------------------------
__global__ void k_attn_spatial(const float* __restrict__ qkv,
                               const int* __restrict__ nei,
                               float* __restrict__ out) {
  __shared__ float pbuf[16 * NN];          // 48 KB
  int head = blockIdx.y;
  int row0 = blockIdx.x * 16;
  int lane = threadIdx.x & 31, wv = threadIdx.x >> 5;
  int m = lane & 15;
  int kk = (lane >> 4) * 2;                // f32 16x16x4 A/B striping
  const float scale = 0.5f;                // 1/sqrt(4)
  v2f aq;
  aq[0] = qkv[(size_t)(row0 + m) * 96 + head * 4 + kk] * scale;
  aq[1] = qkv[(size_t)(row0 + m) * 96 + head * 4 + kk + 1] * scale;
  int nn = lane & 15, mb = (lane >> 4) << 3;
  for (int jt = wv; jt < NN / 16; jt += 8) {
    int col0 = jt * 16;
    v2f bk;
    bk[0] = qkv[(size_t)(col0 + nn) * 96 + 32 + head * 4 + kk];
    bk[1] = qkv[(size_t)(col0 + nn) * 96 + 32 + head * 4 + kk + 1];
    v8f s = {};
    s = __builtin_amdgcn_wmma_f32_16x16x4_f32(false, aq, false, bk, (short)0, s,
                                              false, false);
#pragma unroll
    for (int v = 0; v < 8; ++v) {
      int i = row0 + mb + v, j = col0 + nn;
      int mv = nei[(size_t)i * NN + j] + (i == j ? 1 : 0);
      float add = (mv == 0) ? -1e20f : ((mv == 1) ? 0.0f : (float)mv);
      pbuf[(mb + v) * NN + j] = s[v] + add;
    }
  }
  __syncthreads();
  for (int r = wv; r < 16; r += 8) {       // softmax: one wave per row
    float mx = -1e30f;
    for (int j = lane; j < NN; j += 32) mx = fmaxf(mx, pbuf[r * NN + j]);
    for (int off = 16; off; off >>= 1) mx = fmaxf(mx, __shfl_xor(mx, off, 32));
    float sum = 0.0f;
    for (int j = lane; j < NN; j += 32) {
      float e = __expf(pbuf[r * NN + j] - mx);
      pbuf[r * NN + j] = e;
      sum += e;
    }
    for (int off = 16; off; off >>= 1) sum += __shfl_xor(sum, off, 32);
    float inv = 1.0f / sum;
    for (int j = lane; j < NN; j += 32) pbuf[r * NN + j] *= inv;
  }
  __syncthreads();
  for (int r = wv; r < 16; r += 8) {       // O = P @ V  (4 cols/head)
    int c = lane & 3, g = lane >> 2;       // 8 lane-groups split the K=768 sum
    float acc = 0.0f;
    for (int j = g; j < NN; j += 8)
      acc += pbuf[r * NN + j] * qkv[(size_t)j * 96 + 64 + head * 4 + c];
    acc += __shfl_xor(acc, 4, 32);
    acc += __shfl_xor(acc, 8, 32);
    acc += __shfl_xor(acc, 16, 32);
    if (g == 0) out[(size_t)(row0 + r) * 32 + head * 4 + c] = acc;
  }
}

// ---------------------------------------------------------------------------
// Temporal attention (L=8, no mask): one thread per (b, head, query row).
// qkv rows indexed (l*B + b).
// ---------------------------------------------------------------------------
__global__ void k_attn_temporal(const float* __restrict__ qkv,
                                float* __restrict__ out, int B) {
  int idx = blockIdx.x * blockDim.x + threadIdx.x;
  if (idx >= B * NHEADS * TT) return;
  int i = idx & 7, h = (idx >> 3) & 7, b = idx >> 6;
  float q[4];
#pragma unroll
  for (int d = 0; d < 4; ++d) q[d] = 0.5f * qkv[(size_t)(i * B + b) * 96 + h * 4 + d];
  float s[8];
  float mx = -1e30f;
#pragma unroll
  for (int j = 0; j < 8; ++j) {
    float t = 0.0f;
#pragma unroll
    for (int d = 0; d < 4; ++d)
      t += q[d] * qkv[(size_t)(j * B + b) * 96 + 32 + h * 4 + d];
    s[j] = t;
    mx = fmaxf(mx, t);
  }
  float sum = 0.0f, o[4] = {0, 0, 0, 0};
#pragma unroll
  for (int j = 0; j < 8; ++j) {
    float p = __expf(s[j] - mx);
    sum += p;
#pragma unroll
    for (int d = 0; d < 4; ++d)
      o[d] += p * qkv[(size_t)(j * B + b) * 96 + 64 + h * 4 + d];
  }
  float inv = 1.0f / sum;
#pragma unroll
  for (int d = 0; d < 4; ++d) out[(size_t)(i * B + b) * 32 + h * 4 + d] = o[d] * inv;
}

// out = LayerNorm(x + delta) over 32 features; one wave32 per row.
__global__ void k_ln_res(const float* __restrict__ x, const float* __restrict__ dl,
                         const float* __restrict__ g, const float* __restrict__ b,
                         float* __restrict__ out, int M) {
  int lane = threadIdx.x & 31, wv = threadIdx.x >> 5;
  int row = blockIdx.x * 8 + wv;
  if (row >= M) return;
  float v = x[(size_t)row * 32 + lane] + dl[(size_t)row * 32 + lane];
  float s = v;
  for (int off = 16; off; off >>= 1) s += __shfl_xor(s, off, 32);
  float mean = s * (1.0f / 32.0f);
  float c = v - mean;
  float q = c * c;
  for (int off = 16; off; off >>= 1) q += __shfl_xor(q, off, 32);
  out[(size_t)row * 32 + lane] = c * rsqrtf(q * (1.0f / 32.0f) + 1e-5f) * g[lane] + b[lane];
}

// relu(pos_fc(concat(emb(src2), pe(t)))) ; one thread per output element.
__global__ void k_embed_pos(const float* __restrict__ src, const float* __restrict__ We,
                            const float* __restrict__ Be, const float* __restrict__ Wp,
                            const float* __restrict__ Bp, float* __restrict__ out) {
  int idx = blockIdx.x * blockDim.x + threadIdx.x;
  if (idx >= TT * NN * DD) return;
  int o = idx & 31, row = idx >> 5;
  int t = row / NN;
  float s0 = src[(size_t)row * 2], s1 = src[(size_t)row * 2 + 1];
  float acc = Bp[o];
#pragma unroll 8
  for (int j = 0; j < 32; ++j) {
    float e = s0 * We[j * 2] + s1 * We[j * 2 + 1] + Be[j];
    acc += e * Wp[o * 64 + j];
  }
  const float cln = -0.28782313662425572f;  // -ln(10000)/32
  for (int j = 0; j < 32; j += 2) {
    float ang = (float)t * __expf(cln * (float)j);
    acc += __sinf(ang) * Wp[o * 64 + 32 + j];
    acc += __cosf(ang) * Wp[o * 64 + 32 + j + 1];
  }
  out[idx] = fmaxf(acc, 0.0f);
}

// semantic attention pieces
__global__ void k_sem_score(const float* __restrict__ z, const float* __restrict__ w1,
                            const float* __restrict__ b1, const float* __restrict__ w2,
                            float* __restrict__ sc) {
  int idx = blockIdx.x * blockDim.x + threadIdx.x;
  if (idx >= NN * RR) return;
  int r = idx % RR, n = idx / RR;
  const float* zp = z + (size_t)n * (RR * 32) + r * 32;
  float acc = 0.0f;
  for (int h = 0; h < 128; ++h) {
    float t = b1[h];
#pragma unroll 8
    for (int d = 0; d < 32; ++d) t += zp[d] * w1[h * 32 + d];
    acc += w2[h] * tanhf(t);
  }
  sc[idx] = acc;
}

__global__ void k_sem_reduce(const float* __restrict__ sc, float* __restrict__ beta) {
  __shared__ float sm[RR];
  int lane = threadIdx.x & 31, wv = threadIdx.x >> 5;
  if (wv < RR) {
    float s = 0.0f;
    for (int n = lane; n < NN; n += 32) s += sc[n * RR + wv];
    for (int off = 16; off; off >>= 1) s += __shfl_xor(s, off, 32);
    if (lane == 0) sm[wv] = s * (1.0f / (float)NN);
  }
  __syncthreads();
  if (threadIdx.x == 0) {
    float m = fmaxf(sm[0], fmaxf(sm[1], sm[2]));
    float e0 = __expf(sm[0] - m), e1 = __expf(sm[1] - m), e2 = __expf(sm[2] - m);
    float inv = 1.0f / (e0 + e1 + e2);
    beta[0] = e0 * inv; beta[1] = e1 * inv; beta[2] = e2 * inv;
  }
}

__global__ void k_sem_combine(const float* __restrict__ z, const float* __restrict__ beta,
                              float* __restrict__ out) {
  int idx = blockIdx.x * blockDim.x + threadIdx.x;
  if (idx >= NN * 32) return;
  int n = idx >> 5, d = idx & 31;
  const float* zp = z + (size_t)n * (RR * 32) + d;
  out[idx] = beta[0] * zp[0] + beta[1] * zp[32] + beta[2] * zp[64];
}

__global__ void k_pack_z(const float* __restrict__ src, float* __restrict__ dst, int r) {
  int idx = blockIdx.x * blockDim.x + threadIdx.x;
  if (idx >= NN * 32) return;
  int n = idx >> 5, d = idx & 31;
  dst[(size_t)n * (RR * 32) + r * 32 + d] = src[idx];
}

__global__ void k_concat(const float* __restrict__ a, const float* __restrict__ b,
                         float* __restrict__ f) {
  int idx = blockIdx.x * blockDim.x + threadIdx.x;
  if (idx >= NN * 64) return;
  int n = idx >> 6, c = idx & 63;
  f[idx] = (c < 32) ? a[(size_t)n * 32 + c] : b[(size_t)n * 32 + c - 32];
}

// ---------------------------------------------------------------------------
// Host side
// ---------------------------------------------------------------------------
struct LayerP {
  const float *in_w, *in_b, *out_w, *out_b, *l1w, *l1b, *l2w, *l2b;
  const float *n1g, *n1b, *n2g, *n2b;
};
struct LayerSW {           // pre-swizzled f16 weights
  _Float16 *in_w, *out_w, *l1w, *l2w;
};

static LayerP layer_at(void* const* d_in, int base) {
  LayerP p;
  p.in_b  = (const float*)d_in[base + 0];
  p.in_w  = (const float*)d_in[base + 1];
  p.out_b = (const float*)d_in[base + 2];
  p.out_w = (const float*)d_in[base + 3];
  p.l1b   = (const float*)d_in[base + 4];
  p.l1w   = (const float*)d_in[base + 5];
  p.l2b   = (const float*)d_in[base + 6];
  p.l2w   = (const float*)d_in[base + 7];
  p.n1b   = (const float*)d_in[base + 8];
  p.n1g   = (const float*)d_in[base + 9];
  p.n2b   = (const float*)d_in[base + 10];
  p.n2g   = (const float*)d_in[base + 11];
  return p;
}

static void run_swz(const float* W, _Float16* out, int N, int K, hipStream_t s) {
  int total = (N / 16) * (K / 32) * 32;
  k_swz<<<(total + 255) / 256, 256, 0, s>>>(W, out, N, K);
}

static void run_gemm(const float* A, const _Float16* Ws, const float* bias, float* C,
                     int M, int N, int K, hipStream_t s) {
  int tiles = (M / 16) * (N / 16);
  k_gemm<0><<<(tiles + 3) / 4, 128, 0, s>>>(A, Ws, bias, C, M, N, K);
}

static void run_enc_layer(const LayerP& P, const LayerSW& S, float* x, int L, int B,
                          const int* neim, float* qkv, float* abuf, float* tbuf,
                          hipStream_t s) {
  int M = L * B;
  run_gemm(x, S.in_w, P.in_b, qkv, M, 96, 32, s);
  if (neim) {
    k_attn_spatial<<<dim3(NN / 16, NHEADS), 256, 0, s>>>(qkv, neim, abuf);
  } else {
    int th = B * NHEADS * TT;
    k_attn_temporal<<<(th + 255) / 256, 256, 0, s>>>(qkv, abuf, B);
  }
  run_gemm(abuf, S.out_w, P.out_b, tbuf, M, 32, 32, s);
  k_ln_res<<<(M + 7) / 8, 256, 0, s>>>(x, tbuf, P.n1g, P.n1b, x, M);
  k_ffn<<<(M / 16 + 3) / 4, 128, 0, s>>>(x, S.l1w, P.l1b, S.l2w, P.l2b, tbuf, M);
  k_ln_res<<<(M + 7) / 8, 256, 0, s>>>(x, tbuf, P.n2g, P.n2b, x, M);
}

static void run_semantic(const float* z, const float* w1, const float* b1,
                         const float* w2, float* score, float* beta, float* out,
                         hipStream_t s) {
  k_sem_score<<<(NN * RR + 127) / 128, 128, 0, s>>>(z, w1, b1, w2, score);
  k_sem_reduce<<<1, 128, 0, s>>>(score, beta);
  k_sem_combine<<<(NN * 32 + 255) / 256, 256, 0, s>>>(z, beta, out);
}

extern "C" void kernel_launch(void* const* d_in, const int* in_sizes, int n_in,
                              void* d_out, int out_size, void* d_ws, size_t ws_size,
                              hipStream_t stream) {
  (void)in_sizes; (void)n_in; (void)out_size; (void)ws_size;
  const float* nodes_cur = (const float*)d_in[0];
  const float* nodes_abs = (const float*)d_in[1];
  const int*   nei       = (const int*)d_in[2];
  // params pytree (jax flatten: dict keys sorted at every level), base = 3
  const float* emb_s_b = (const float*)d_in[3];
  const float* emb_s_w = (const float*)d_in[4];
  const float* emb_t_b = (const float*)d_in[5];
  const float* emb_t_w = (const float*)d_in[6];
  const float* fus_b   = (const float*)d_in[7];
  const float* fus_w   = (const float*)d_in[8];
  const float* p1_b    = (const float*)d_in[9];
  const float* p1_w    = (const float*)d_in[10];
  const float* p2_b    = (const float*)d_in[11];
  const float* p2_w    = (const float*)d_in[12];
  LayerP lp[6];
  lp[0] = layer_at(d_in, 13);  // se1_0
  lp[1] = layer_at(d_in, 25);  // se1_1
  lp[2] = layer_at(d_in, 37);  // se2_0
  lp[3] = layer_at(d_in, 49);  // se2_1
  lp[4] = layer_at(d_in, 67);  // te1
  lp[5] = layer_at(d_in, 79);  // te2
  const float* sem1_b1 = (const float*)d_in[61];
  const float* sem1_w1 = (const float*)d_in[62];
  const float* sem1_w2 = (const float*)d_in[63];
  const float* sem2_b1 = (const float*)d_in[64];
  const float* sem2_w1 = (const float*)d_in[65];
  const float* sem2_w2 = (const float*)d_in[66];

  // ---- workspace bump allocation: f32 region, then f16 swizzled weights
  float* w = (float*)d_ws;
  float* xbuf  = w; w += TT * NN * DD;          // 6144x32 (x / t1 / y)
  float* qkv   = w; w += TT * NN * 96;          // 6144x96
  float* abuf  = w; w += TT * NN * DD;
  float* tbuf  = w; w += TT * NN * DD;
  float* spx   = w; w += NN * DD;               // spatial working x
  float* zbuf  = w; w += NN * RR * DD;          // (N, R, D)
  float* sfeat = w; w += NN * DD;
  float* sp2   = w; w += TT * NN * DD;
  float* fbuf  = w; w += NN * 64;
  float* sscore = w; w += NN * RR;
  float* sbeta  = w; w += 16;

  _Float16* hw = (_Float16*)w;
  LayerSW sw[6];
  for (int l = 0; l < 6; ++l) {
    sw[l].in_w  = hw; hw += 96 * 32;            // 6 frags  * 512
    sw[l].out_w = hw; hw += 32 * 32;            // 2 frags  * 512
    sw[l].l1w   = hw; hw += HID * 32;           // 128 frags* 512
    sw[l].l2w   = hw; hw += 32 * HID;           // 128 frags* 512
  }
  _Float16* fus_sw = hw; hw += 32 * 64;         // 4 frags * 512

  // ---- one-time (per launch) weight swizzle to fragment-ordered f16
  for (int l = 0; l < 6; ++l) {
    run_swz(lp[l].in_w,  sw[l].in_w,  96, 32, stream);
    run_swz(lp[l].out_w, sw[l].out_w, 32, 32, stream);
    run_swz(lp[l].l1w,   sw[l].l1w,   HID, 32, stream);
    run_swz(lp[l].l2w,   sw[l].l2w,   32, HID, stream);
  }
  run_swz(fus_w, fus_sw, 32, 64, stream);

  const size_t frame_f = (size_t)NN * DD * sizeof(float);
  const int eg = (TT * NN * DD + 255) / 256;

  // ---- temporal branch 1: x = relu(pos1(concat(emb_t(cur), pe))) ; te1
  k_embed_pos<<<eg, 256, 0, stream>>>(nodes_cur, emb_t_w, emb_t_b, p1_w, p1_b, xbuf);
  run_enc_layer(lp[4], sw[4], xbuf, TT, NN, nullptr, qkv, abuf, tbuf, stream);

  // ---- spatial se1 on last frame, 3 masks (frame T-1)
  for (int r = 0; r < RR; ++r) {
    hipMemcpyAsync(spx, xbuf + (size_t)(TT - 1) * NN * DD, frame_f,
                   hipMemcpyDeviceToDevice, stream);
    const int* mr = nei + ((size_t)r * TT + (TT - 1)) * NN * NN;
    run_enc_layer(lp[0], sw[0], spx, NN, 1, mr, qkv, abuf, tbuf, stream);
    run_enc_layer(lp[1], sw[1], spx, NN, 1, mr, qkv, abuf, tbuf, stream);
    k_pack_z<<<(NN * 32 + 255) / 256, 256, 0, stream>>>(spx, zbuf, r);
  }
  run_semantic(zbuf, sem1_w1, sem1_b1, sem1_w2, sscore, sbeta, sfeat, stream);

  // ---- second branch: y = relu(pos2(concat(emb_s(abs), pe)))
  k_embed_pos<<<eg, 256, 0, stream>>>(nodes_abs, emb_s_w, emb_s_b, p2_w, p2_b, xbuf);

  // ---- se2 per frame, 3 masks each, semantic -> sp2[t]
  for (int t = 0; t < TT; ++t) {
    for (int r = 0; r < RR; ++r) {
      hipMemcpyAsync(spx, xbuf + (size_t)t * NN * DD, frame_f,
                     hipMemcpyDeviceToDevice, stream);
      const int* mr = nei + ((size_t)r * TT + t) * NN * NN;
      run_enc_layer(lp[2], sw[2], spx, NN, 1, mr, qkv, abuf, tbuf, stream);
      run_enc_layer(lp[3], sw[3], spx, NN, 1, mr, qkv, abuf, tbuf, stream);
      k_pack_z<<<(NN * 32 + 255) / 256, 256, 0, stream>>>(spx, zbuf, r);
    }
    run_semantic(zbuf, sem2_w1, sem2_b1, sem2_w2, sscore, sbeta,
                 sp2 + (size_t)t * NN * DD, stream);
  }

  // ---- te2 on sp2, take last frame
  run_enc_layer(lp[5], sw[5], sp2, TT, NN, nullptr, qkv, abuf, tbuf, stream);

  // ---- fusion
  k_concat<<<(NN * 64 + 255) / 256, 256, 0, stream>>>(
      sfeat, sp2 + (size_t)(TT - 1) * NN * DD, fbuf);
  run_gemm(fbuf, fus_sw, fus_b, (float*)d_out, NN, 32, 64, stream);
}
